// ScOTLayer_51350628991014
// MI455X (gfx1250) — matrix operand order repaired and checked by reading, not compiled, not run on hardware
//
#include <hip/hip_runtime.h>
#include <hip/hip_bf16.h>
#include <math.h>

typedef __attribute__((ext_vector_type(16))) _Float16 v16h;
typedef __attribute__((ext_vector_type(8)))  _Float16 v8h;
typedef __attribute__((ext_vector_type(4)))  _Float16 v4h;
typedef __attribute__((ext_vector_type(8)))  float    v8f;

#define DEVINL __device__ __forceinline__

// ---------------- problem constants ----------------
constexpr int C_   = 96;     // channels
constexpr int NH_  = 3;      // heads
constexpr int HD_  = 32;     // head dim
constexpr int N_   = 64;     // tokens per window
constexpr int IMG  = 256;    // H == W
constexpr int SSH  = 4;      // shift size
constexpr int TOK  = 2 * IMG * IMG;   // 131072 tokens total
constexpr int NWIN = 2048;            // total windows
constexpr int MLP_ = 384;

// ---------------- workspace layout (bytes) ----------------
// all weights stored TRANSPOSED ([N][K]) as f16 so B-fragments are contiguous
constexpr size_t WS_BIAS = 0;                         // 3*64*64 f32  = 49152
constexpr size_t WS_QW   = 49152;                     // 96*96 f16 (transposed)
constexpr size_t WS_KW   = WS_QW  + 18432;
constexpr size_t WS_VW   = WS_KW  + 18432;
constexpr size_t WS_PW   = WS_VW  + 18432;
constexpr size_t WS_FC1  = WS_PW  + 18432;            // [384][96] f16
constexpr size_t WS_FC2  = WS_FC1 + 73728;            // [96][384] f16
constexpr size_t WS_HS   = WS_FC2 + 73728;            // 131072*96 f32

// ================= WMMA fragment helpers (doc layouts, wave32) =================
// A (16x32 f16, row-major src, stride lda MULTIPLE OF 8, 16B-aligned base).
// Per lane: two contiguous 8-half runs -> two 16B LDS loads.
DEVINL v16h frag_a(const _Float16* a, int lda, int lane) {
    const _Float16* row = a + (lane & 15) * lda + ((lane < 16) ? 0 : 8);
    v8h lo = *(const v8h*)(row);
    v8h hi = *(const v8h*)(row + 16);
    return __builtin_shufflevector(lo, hi, 0, 1, 2, 3, 4, 5, 6, 7,
                                   8, 9, 10, 11, 12, 13, 14, 15);
}

// B = X^T where X is row-major [N][K] with stride ldr (MULTIPLE OF 8, aligned).
// -> two contiguous 8-half runs per lane (same k order as frag_a) = two 16B loads.
DEVINL v16h frag_bt(const _Float16* x, int ldr, int lane) {
    const _Float16* row = x + (lane & 15) * ldr + ((lane < 16) ? 0 : 8);
    v8h lo = *(const v8h*)(row);
    v8h hi = *(const v8h*)(row + 16);
    return __builtin_shufflevector(lo, hi, 0, 1, 2, 3, 4, 5, 6, 7,
                                   8, 9, 10, 11, 12, 13, 14, 15);
}

DEVINL v8f wmma_f16(v16h a, v16h b, v8f c) {
    return __builtin_amdgcn_wmma_f32_16x16x32_f16(false, a, false, b,
                                                  (short)0, c, false, false);
}

// C store (16x16 f32): element r -> M = r + (lane<16?0:8), N = lane&15.
DEVINL void store_c_f16(v8f acc, _Float16* dst, int ldd, int lane, float bias) {
    const int n    = lane & 15;
    const int moff = (lane < 16) ? 0 : 8;
#pragma unroll
    for (int r = 0; r < 8; ++r)
        dst[(moff + r) * ldd + n] = (_Float16)(acc[r] + bias);
}
// transposed C store: dst[N][M], 8 contiguous halfs per lane = one 16B store.
DEVINL void store_c_f16_T(v8f acc, _Float16* dst, int ldd, int lane, float bias) {
    const int n    = lane & 15;
    const int moff = (lane < 16) ? 0 : 8;
    v8h t;
#pragma unroll
    for (int r = 0; r < 8; ++r) t[r] = (_Float16)(acc[r] + bias);
    *(v8h*)(dst + n * ldd + moff) = t;
}
DEVINL void store_c_f32(v8f acc, float* dst, int ldd, int lane, float bias) {
    const int n    = lane & 15;
    const int moff = (lane < 16) ? 0 : 8;
#pragma unroll
    for (int r = 0; r < 8; ++r)
        dst[(moff + r) * ldd + n] = acc[r] + bias;
}

// ================= kernel 1: weights -> f16, transposed =================
__global__ void wconv_kernel(const float* __restrict__ qw, const float* __restrict__ kw,
                             const float* __restrict__ vw, const float* __restrict__ pw,
                             const float* __restrict__ f1, const float* __restrict__ f2,
                             _Float16* dq, _Float16* dk, _Float16* dv, _Float16* dp,
                             _Float16* d1, _Float16* d2) {
    int i = blockIdx.x * blockDim.x + threadIdx.x;
    if (i < 9216) {                       // [96][96] -> transpose
        int k = i / 96, n = i % 96;
        dq[n * 96 + k] = (_Float16)qw[i];
        dk[n * 96 + k] = (_Float16)kw[i];
        dv[n * 96 + k] = (_Float16)vw[i];
        dp[n * 96 + k] = (_Float16)pw[i];
    }
    if (i < 36864) {
        int k1 = i / 384, n1 = i % 384;   // fc1 [96][384] -> [384][96]
        d1[n1 * 96 + k1] = (_Float16)f1[i];
        int k2 = i / 96, n2 = i % 96;     // fc2 [384][96] -> [96][384]
        d2[n2 * 384 + k2] = (_Float16)f2[i];
    }
}

// ================= kernel 2: continuous position bias table =================
// stored TRANSPOSED: bias[h][j][i]  (j = key, i = query) so that the S-tile
// epilogue reads 8 contiguous i's per lane -> two float4 loads per tile.
DEVINL float cpb_coord(int d) {
    float t = (float)d * (8.0f / 7.0f);
    float s = (t > 0.f) ? 1.f : ((t < 0.f) ? -1.f : 0.f);
    return s * log2f(fabsf(t) + 1.f) * 0.333333333f;   // /log2(8)
}

__global__ void cpb_kernel(const float* __restrict__ w1, const float* __restrict__ b1,
                           const float* __restrict__ w2, float* __restrict__ bias) {
    int gid = blockIdx.x * blockDim.x + threadIdx.x;   // 4096 (i,j) pairs
    if (gid >= N_ * N_) return;
    int i = gid >> 6, j = gid & 63;
    int dy = (i >> 3) - (j >> 3);
    int dx = (i & 7) - (j & 7);
    float c0 = cpb_coord(dy), c1 = cpb_coord(dx);
    float a0 = 0.f, a1 = 0.f, a2 = 0.f;
    for (int h = 0; h < 512; ++h) {
        float v = fmaxf(c0 * w1[h] + c1 * w1[512 + h] + b1[h], 0.f);
        a0 += v * w2[h * 3 + 0];
        a1 += v * w2[h * 3 + 1];
        a2 += v * w2[h * 3 + 2];
    }
    int tjd = j * 64 + i;                 // transposed store
    bias[0 * 4096 + tjd] = 16.f / (1.f + expf(-a0));
    bias[1 * 4096 + tjd] = 16.f / (1.f + expf(-a1));
    bias[2 * 4096 + tjd] = 16.f / (1.f + expf(-a2));
}

// ================= kernel 3: fused shifted-window attention =================
constexpr int XS  = 104;   // xw f16 row stride   (mult of 8)
constexpr int SU  = 68;    // S f32 row stride (union region)
constexpr int PS  = 136;   // P f16 row stride = 2*SU (mult of 8)
constexpr int QS  = 104;   // q/k f16 row stride  (mult of 8)
constexpr int VTS = 72;    // v^T f16 row stride  (mult of 8)
constexpr int OS  = 100;   // proj-out f32 row stride (overlay on k+vT)
constexpr int OFF_UNION = 0;                 // max(64*XS*2, 64*SU*4) = 17408
constexpr int OFF_Q  = 17408;                // 64*QS*2  = 13312
constexpr int OFF_K  = OFF_Q + 13312;
constexpr int OFF_VT = OFF_K + 13312;        // 96*VTS*2 = 13824
constexpr int SMEM_ATTN = OFF_VT + 13824;    // 57856 B

DEVINL int region_label(int wy, int wx, int tok) {
    int rr = wy * 8 + (tok >> 3);
    int cc = wx * 8 + (tok & 7);
    int a = (rr < IMG - 8) ? 0 : ((rr < IMG - SSH) ? 1 : 2);
    int b = (cc < IMG - 8) ? 0 : ((cc < IMG - SSH) ? 1 : 2);
    return a * 3 + b;
}

__global__ void __launch_bounds__(256)
attn_kernel(const float* __restrict__ hid, const float* __restrict__ timev,
            const float* __restrict__ q_b, const float* __restrict__ v_b,
            const float* __restrict__ p_b, const float* __restrict__ lscale,
            const float* __restrict__ bias_tab,
            const _Float16* __restrict__ qwT, const _Float16* __restrict__ kwT,
            const _Float16* __restrict__ vwT, const _Float16* __restrict__ pwT,
            const float* __restrict__ ln_ww, const float* __restrict__ ln_wb,
            const float* __restrict__ ln_bw, const float* __restrict__ ln_bb,
            float* __restrict__ hs_out) {
    __shared__ __align__(16) char smem[SMEM_ATTN];
    _Float16* xw16 = (_Float16*)(smem + OFF_UNION);
    float*    s32  = (float*)(smem + OFF_UNION);
    _Float16* q16  = (_Float16*)(smem + OFF_Q);
    _Float16* k16  = (_Float16*)(smem + OFF_K);
    _Float16* vT16 = (_Float16*)(smem + OFF_VT);       // [96 ch][64 tok]
    float*    o32  = (float*)(smem + OFF_K);           // overlays k+vT after heads

    const int tid  = threadIdx.x;
    const int wave = tid >> 5;
    const int lane = tid & 31;
    const int blk  = blockIdx.x;
    const int bImg = blk >> 10;
    const int wy   = (blk >> 5) & 31;
    const int wx   = blk & 31;

    // ---- phase 0: shifted-window gather -> LDS f16 (64 tok x 96 ch) ----
    for (int idx4 = tid; idx4 < N_ * 24; idx4 += 256) {
        int t  = idx4 / 24;
        int c4 = idx4 % 24;
        int ys = (wy * 8 + (t >> 3) + SSH) & 255;
        int xs = (wx * 8 + (t & 7) + SSH) & 255;
        const float4 f = *reinterpret_cast<const float4*>(
            hid + ((size_t)(bImg * IMG * IMG + ys * IMG + xs)) * C_ + c4 * 4);
        v4h h4;
        h4[0] = (_Float16)f.x; h4[1] = (_Float16)f.y;
        h4[2] = (_Float16)f.z; h4[3] = (_Float16)f.w;
        *(v4h*)(xw16 + t * XS + c4 * 4) = h4;
    }
    __syncthreads();

    // ---- phase 1: QKV GEMMs, B-fragments hoisted over 4 M-tiles ----
    for (int pp = wave; pp < 18; pp += 8) {
        int proj = pp / 6, nt = pp % 6;
        const _Float16* W = (proj == 0) ? qwT : (proj == 1) ? kwT : vwT;
        v8f acc[4] = {};
#pragma unroll
        for (int ks = 0; ks < 3; ++ks) {
            v16h b = frag_bt(W + (nt * 16) * C_ + ks * 32, C_, lane);
#pragma unroll
            for (int mt = 0; mt < 4; ++mt) {
                v16h a = frag_a(xw16 + mt * 16 * XS + ks * 32, XS, lane);
                acc[mt] = wmma_f16(a, b, acc[mt]);
            }
        }
        int col = nt * 16 + (lane & 15);
        if (proj == 0) {
            float bq = q_b[col];
#pragma unroll
            for (int mt = 0; mt < 4; ++mt)
                store_c_f16(acc[mt], q16 + mt * 16 * QS + nt * 16, QS, lane, bq);
        } else if (proj == 1) {
#pragma unroll
            for (int mt = 0; mt < 4; ++mt)
                store_c_f16(acc[mt], k16 + mt * 16 * QS + nt * 16, QS, lane, 0.f);
        } else {  // v stored TRANSPOSED: [ch][tok], one 16B store per lane
            float bv = v_b[col];
#pragma unroll
            for (int mt = 0; mt < 4; ++mt)
                store_c_f16_T(acc[mt], vT16 + (nt * 16) * VTS + mt * 16, VTS, lane, bv);
        }
    }
    __syncthreads();

    // ---- phase 2: per-head cosine attention ----
    for (int h = 0; h < NH_; ++h) {
        // 2a: row normalization (fold logit scale into q)
        if (tid < 128) {
            int row = tid & 63;
            _Float16* base = ((tid < 64) ? q16 : k16) + row * QS + 32 * h;
            float ss = 0.f;
#pragma unroll
            for (int d = 0; d < HD_; ++d) { float v = (float)base[d]; ss += v * v; }
            float inv = 1.f / fmaxf(sqrtf(ss), 1e-12f);
            if (tid < 64) inv *= expf(fminf(lscale[h], 4.6051702f));  // log(100)
#pragma unroll
            for (int d = 0; d < HD_; ++d) base[d] = (_Float16)((float)base[d] * inv);
        }
        __syncthreads();

        // 2b: S = qn @ kn^T (16 tiles, K=32) + bias + shift mask -> s32
        for (int t2 = 0; t2 < 2; ++t2) {
            int tile = wave * 2 + t2;
            int mt = tile >> 2, nt = tile & 3;
            v16h a = frag_a(q16 + mt * 16 * QS + 32 * h, QS, lane);
            v16h b = frag_bt(k16 + nt * 16 * QS + 32 * h, QS, lane);
            v8f acc = {};
            acc = wmma_f16(a, b, acc);
            int n = lane & 15, moff = (lane < 16) ? 0 : 8;
            int j = nt * 16 + n;
            int lblj = region_label(wy, wx, j);
            // transposed bias: 8 contiguous i's per lane -> two float4 loads
            const float* bcol = bias_tab + h * 4096 + j * 64 + mt * 16 + moff;
            float4 b0 = *(const float4*)(bcol);
            float4 b1 = *(const float4*)(bcol + 4);
            float bv[8] = {b0.x, b0.y, b0.z, b0.w, b1.x, b1.y, b1.z, b1.w};
#pragma unroll
            for (int r = 0; r < 8; ++r) {
                int i = mt * 16 + moff + r;
                float m = (region_label(wy, wx, i) == lblj) ? 0.f : -100.f;
                s32[i * SU + j] = acc[r] + bv[r] + m;
            }
        }
        __syncthreads();

        // 2c: row softmax; overwrite s row (f32) with p row (f16) in place
        if (tid < 64) {
            float*    srow = s32 + tid * SU;
            _Float16* prow = (_Float16*)srow;            // same bytes, fwd-safe
            float mx = -1e30f;
            for (int j = 0; j < 64; ++j) mx = fmaxf(mx, srow[j]);
            float sum = 0.f;
            for (int j = 0; j < 64; ++j) {
                float e = expf(srow[j] - mx);            // read before clobber
                sum += e;
                prow[j] = (_Float16)e;
            }
            float inv = 1.f / sum;
            for (int j = 0; j < 64; ++j) prow[j] = (_Float16)((float)prow[j] * inv);
        }
        __syncthreads();

        // 2d: ctx = P @ v (8 tiles, K=64), B from v^T -> contiguous loads;
        //     write ctx into q's head-h columns
        {
            _Float16* p16 = (_Float16*)s32;
            int mt = wave >> 1, nt = wave & 1;
            v8f acc = {};
#pragma unroll
            for (int ks = 0; ks < 2; ++ks) {
                v16h a = frag_a(p16 + mt * 16 * PS + ks * 32, PS, lane);
                v16h b = frag_bt(vT16 + (32 * h + nt * 16) * VTS + ks * 32, VTS, lane);
                acc = wmma_f16(a, b, acc);
            }
            store_c_f16(acc, q16 + mt * 16 * QS + 32 * h + nt * 16, QS, lane, 0.f);
        }
        __syncthreads();
    }

    // ---- phase 3: output projection, B hoisted over 4 M-tiles ----
    if (wave < 6) {
        int nt = wave;
        v8f acc[4] = {};
#pragma unroll
        for (int ks = 0; ks < 3; ++ks) {
            v16h b = frag_bt(pwT + (nt * 16) * C_ + ks * 32, C_, lane);
#pragma unroll
            for (int mt = 0; mt < 4; ++mt) {
                v16h a = frag_a(q16 + mt * 16 * QS + ks * 32, QS, lane);
                acc[mt] = wmma_f16(a, b, acc[mt]);
            }
        }
        float bp = p_b[nt * 16 + (lane & 15)];
        __syncthreads();   // all waves: k/vT fully read before o32 overlay
#pragma unroll
        for (int mt = 0; mt < 4; ++mt)
            store_c_f32(acc[mt], o32 + mt * 16 * OS + nt * 16, OS, lane, bp);
    } else {
        __syncthreads();
    }
    __syncthreads();

    // ---- phase 4: window-reverse scatter + conditional LN + residual ----
    if (tid < 64) {
        int t = tid;
        int yd = (wy * 8 + (t >> 3) + SSH) & 255;   // reverse of -SSH roll
        int xd = (wx * 8 + (t & 7) + SSH) & 255;
        size_t tok = (size_t)bImg * IMG * IMG + yd * IMG + xd;
        const float* row = o32 + t * OS;
        float mean = 0.f, msq = 0.f;
#pragma unroll
        for (int c4 = 0; c4 < 24; ++c4) {
            float4 v = *(const float4*)(row + c4 * 4);
            mean += v.x + v.y + v.z + v.w;
            msq  += v.x * v.x + v.y * v.y + v.z * v.z + v.w * v.w;
        }
        mean *= (1.f / C_); msq *= (1.f / C_);
        float rstd = rsqrtf(fmaxf(msq - mean * mean, 0.f) + 1e-5f);
        float tv = timev[bImg];
#pragma unroll 2
        for (int c4 = 0; c4 < 24; ++c4) {
            float4 v  = *(const float4*)(row + c4 * 4);
            float4 sc = *(const float4*)(hid + tok * C_ + c4 * 4);
            float4 ww = *(const float4*)(ln_ww + c4 * 4);
            float4 wb = *(const float4*)(ln_wb + c4 * 4);
            float4 bw = *(const float4*)(ln_bw + c4 * 4);
            float4 bb = *(const float4*)(ln_bb + c4 * 4);
            float4 o;
            o.x = sc.x + (tv * ww.x + wb.x) * ((v.x - mean) * rstd) + (tv * bw.x + bb.x);
            o.y = sc.y + (tv * ww.y + wb.y) * ((v.y - mean) * rstd) + (tv * bw.y + bb.y);
            o.z = sc.z + (tv * ww.z + wb.z) * ((v.z - mean) * rstd) + (tv * bw.z + bb.z);
            o.w = sc.w + (tv * ww.w + wb.w) * ((v.w - mean) * rstd) + (tv * bw.w + bb.w);
            *(float4*)(hs_out + tok * C_ + c4 * 4) = o;
        }
    }
}

// ================= kernel 4: fused MLP (fc1+GELU+fc2) + CLN + residual =======
// 64-token tile, 256 threads; mlp-out (f16) overlays hs16 (dead after fc1).
constexpr int HSS = 104;   // hs f16 stride   (mult of 8)
constexpr int ISS = 392;   // inter f16 stride (mult of 8)
constexpr int MOS = 100;   // mlp-out f16 stride (mult of 4 -> 8B aligned rows)
constexpr int OFF_HS16 = 0;                 // max(64*HSS, 64*MOS)*2 = 13312
constexpr int OFF_INT  = 13312;             // 64*ISS*2 = 50176
constexpr int SMEM_MLP = OFF_INT + 50176;   // 63488 B

__global__ void __launch_bounds__(256)
mlp_kernel(const float* __restrict__ hs, const float* __restrict__ timev,
           const _Float16* __restrict__ f1T, const float* __restrict__ f1b,
           const _Float16* __restrict__ f2T, const float* __restrict__ f2b,
           const float* __restrict__ ln_ww, const float* __restrict__ ln_wb,
           const float* __restrict__ ln_bw, const float* __restrict__ ln_bb,
           float* __restrict__ out) {
    __shared__ __align__(16) char smem[SMEM_MLP];
    _Float16* hs16 = (_Float16*)(smem + OFF_HS16);
    _Float16* mo16 = (_Float16*)(smem + OFF_HS16);     // overlay (after fc1)
    _Float16* it16 = (_Float16*)(smem + OFF_INT);

    const int tid  = threadIdx.x;
    const int wave = tid >> 5;
    const int lane = tid & 31;
    const size_t tb = (size_t)blockIdx.x * 64;

    // prefetch residual rows (re-read at epilogue) while we compute
    if (tid < 64) __builtin_prefetch(hs + (tb + tid) * C_, 0, 1);

    // load 64-token tile of hs -> f16 LDS
    for (int idx4 = tid; idx4 < 64 * 24; idx4 += 256) {
        int t = idx4 / 24, c4 = idx4 % 24;
        const float4 f = *reinterpret_cast<const float4*>(hs + (tb + t) * C_ + c4 * 4);
        v4h h4;
        h4[0] = (_Float16)f.x; h4[1] = (_Float16)f.y;
        h4[2] = (_Float16)f.z; h4[3] = (_Float16)f.w;
        *(v4h*)(hs16 + t * HSS + c4 * 4) = h4;
    }
    __syncthreads();

    // fc1 + exact GELU: [64x96]@[96x384], B hoisted over 4 M-tiles
    for (int nt = wave; nt < 24; nt += 8) {
        v8f acc[4] = {};
#pragma unroll
        for (int ks = 0; ks < 3; ++ks) {
            v16h b = frag_bt(f1T + (nt * 16) * C_ + ks * 32, C_, lane);
#pragma unroll
            for (int mt = 0; mt < 4; ++mt) {
                v16h a = frag_a(hs16 + mt * 16 * HSS + ks * 32, HSS, lane);
                acc[mt] = wmma_f16(a, b, acc[mt]);
            }
        }
        int n = lane & 15, moff = (lane < 16) ? 0 : 8;
        float bias = f1b[nt * 16 + n];
#pragma unroll
        for (int mt = 0; mt < 4; ++mt)
#pragma unroll
            for (int r = 0; r < 8; ++r) {
                float x = acc[mt][r] + bias;
                float g = 0.5f * x * (1.0f + erff(x * 0.70710678f));
                it16[(mt * 16 + moff + r) * ISS + nt * 16 + n] = (_Float16)g;
            }
    }
    __syncthreads();   // hs16 dead from here; mo16 overlay becomes live

    // fc2: [64x384]@[384x96], K=384, B hoisted over 4 M-tiles
    if (wave < 6) {
        int nt = wave;
        v8f acc[4] = {};
#pragma unroll
        for (int ks = 0; ks < 12; ++ks) {
            v16h b = frag_bt(f2T + (nt * 16) * MLP_ + ks * 32, MLP_, lane);
#pragma unroll
            for (int mt = 0; mt < 4; ++mt) {
                v16h a = frag_a(it16 + mt * 16 * ISS + ks * 32, ISS, lane);
                acc[mt] = wmma_f16(a, b, acc[mt]);
            }
        }
        float bias = f2b[nt * 16 + (lane & 15)];
#pragma unroll
        for (int mt = 0; mt < 4; ++mt)
            store_c_f16(acc[mt], mo16 + mt * 16 * MOS + nt * 16, MOS, lane, bias);
    }
    __syncthreads();

    // conditional LN over mlp_out + residual(hs)
    if (tid < 64) {
        size_t tok = tb + tid;
        int bImg = (int)(tok >> 16);
        float tv = timev[bImg];
        const _Float16* row = mo16 + tid * MOS;
        float mean = 0.f, msq = 0.f;
#pragma unroll
        for (int c4 = 0; c4 < 24; ++c4) {
            v4h v = *(const v4h*)(row + c4 * 4);
#pragma unroll
            for (int e = 0; e < 4; ++e) {
                float f = (float)v[e];
                mean += f; msq += f * f;
            }
        }
        mean *= (1.f / C_); msq *= (1.f / C_);
        float rstd = rsqrtf(fmaxf(msq - mean * mean, 0.f) + 1e-5f);
#pragma unroll 2
        for (int c4 = 0; c4 < 24; ++c4) {
            v4h v = *(const v4h*)(row + c4 * 4);
            float4 sc = *(const float4*)(hs + tok * C_ + c4 * 4);
            float4 ww = *(const float4*)(ln_ww + c4 * 4);
            float4 wb = *(const float4*)(ln_wb + c4 * 4);
            float4 bw = *(const float4*)(ln_bw + c4 * 4);
            float4 bb = *(const float4*)(ln_bb + c4 * 4);
            float4 o;
            o.x = sc.x + (tv * ww.x + wb.x) * (((float)v[0] - mean) * rstd) + (tv * bw.x + bb.x);
            o.y = sc.y + (tv * ww.y + wb.y) * (((float)v[1] - mean) * rstd) + (tv * bw.y + bb.y);
            o.z = sc.z + (tv * ww.z + wb.z) * (((float)v[2] - mean) * rstd) + (tv * bw.z + bb.z);
            o.w = sc.w + (tv * ww.w + wb.w) * (((float)v[3] - mean) * rstd) + (tv * bw.w + bb.w);
            *(float4*)(out + tok * C_ + c4 * 4) = o;
        }
    }
}

// ================= host launch =================
extern "C" void kernel_launch(void* const* d_in, const int* in_sizes, int n_in,
                              void* d_out, int out_size, void* d_ws, size_t ws_size,
                              hipStream_t stream) {
    const float* hid   = (const float*)d_in[0];
    const float* timev = (const float*)d_in[1];
    const float* q_w   = (const float*)d_in[2];
    const float* q_b   = (const float*)d_in[3];
    const float* k_w   = (const float*)d_in[4];
    const float* v_w   = (const float*)d_in[5];
    const float* v_b   = (const float*)d_in[6];
    const float* p_w   = (const float*)d_in[7];
    const float* p_b   = (const float*)d_in[8];
    const float* lsc   = (const float*)d_in[9];
    const float* cpbw1 = (const float*)d_in[10];
    const float* cpbb1 = (const float*)d_in[11];
    const float* cpbw2 = (const float*)d_in[12];
    const float* lnbww = (const float*)d_in[13];
    const float* lnbwb = (const float*)d_in[14];
    const float* lnbbw = (const float*)d_in[15];
    const float* lnbbb = (const float*)d_in[16];
    const float* lnaww = (const float*)d_in[17];
    const float* lnawb = (const float*)d_in[18];
    const float* lnabw = (const float*)d_in[19];
    const float* lnabb = (const float*)d_in[20];
    const float* f1w   = (const float*)d_in[21];
    const float* f1b   = (const float*)d_in[22];
    const float* f2w   = (const float*)d_in[23];
    const float* f2b   = (const float*)d_in[24];

    char* ws = (char*)d_ws;
    float*    bias_tab = (float*)(ws + WS_BIAS);
    _Float16* qwT = (_Float16*)(ws + WS_QW);
    _Float16* kwT = (_Float16*)(ws + WS_KW);
    _Float16* vwT = (_Float16*)(ws + WS_VW);
    _Float16* pwT = (_Float16*)(ws + WS_PW);
    _Float16* f1T = (_Float16*)(ws + WS_FC1);
    _Float16* f2T = (_Float16*)(ws + WS_FC2);
    float*    hsb = (float*)(ws + WS_HS);

    wconv_kernel<<<144, 256, 0, stream>>>(q_w, k_w, v_w, p_w, f1w, f2w,
                                          qwT, kwT, vwT, pwT, f1T, f2T);
    cpb_kernel<<<32, 128, 0, stream>>>(cpbw1, cpbb1, cpbw2, bias_tab);
    attn_kernel<<<NWIN, 256, 0, stream>>>(hid, timev, q_b, v_b, p_b, lsc,
                                          bias_tab, qwT, kwT, vwT, pwT,
                                          lnbww, lnbwb, lnbbw, lnbbb, hsb);
    mlp_kernel<<<TOK / 64, 256, 0, stream>>>(hsb, timev, f1T, f1b, f2T, f2b,
                                             lnaww, lnawb, lnabw, lnabb,
                                             (float*)d_out);
}
